// DPNModel_43387759624602
// MI455X (gfx1250) — matrix operand order repaired and checked by reading, not compiled, not run on hardware
//
#include <hip/hip_runtime.h>
#include <hip/hip_bf16.h>
#include <math.h>

// CDNA5 (gfx1250) DPN forward, phase 4.
// All matrix-shaped work (16 VGG convs via im2col, 9x9 glob conv, 210x210 lin2)
// runs through a wave32 WMMA f16->f32 GEMM (v_wmma_f32_16x16x32_f16).
// Each wave computes a 16x64 output strip (4 N-tiles) sharing one A fragment
// per k-step: cuts HBM traffic on the 205MB fc6 weight stream 4x (the roofline-
// dominant term at 23.3 TB/s) and amortizes each A load over 4 WMMAs.
// Fragment layouts per CDNA5 ISA 7.12.2.

typedef __attribute__((ext_vector_type(16))) _Float16 v16h;
typedef __attribute__((ext_vector_type(8)))  float    v8f;

#define NLAB 21
#define NMIX 10
#define KLOC 50
#define KLOC2 (KLOC * KLOC)

static inline int ceil_to(int x, int a) { return ((x + a - 1) / a) * a; }

// ---------------- pack weights: f32 [M][K] -> f16 [Mp][Kp], zero padded ----
__global__ void k_pack_w(const float* __restrict__ W, _Float16* __restrict__ out,
                         int M, int K, int Mp, int Kp) {
  long long idx = (long long)blockIdx.x * blockDim.x + threadIdx.x;
  long long tot = (long long)Mp * Kp;
  if (idx >= tot) return;
  int k = (int)(idx % Kp);
  int m = (int)(idx / Kp);
  float v = (m < M && k < K) ? W[(long long)m * K + k] : 0.f;
  out[idx] = (_Float16)v;
}

// ---------------- im2col: act f32 [cin][Hin][Win] -> f16 patches [Np][Kp] ---
// Row n = output pixel, k ordered (cin, kh, kw) to match OIHW weight flatten.
__global__ void k_im2col(const float* __restrict__ act, _Float16* __restrict__ out,
                         int cin, int Hin, int Win, int Hout, int Wout,
                         int ks, int dil, int pad, int Kp, int Np) {
  long long idx = (long long)blockIdx.x * blockDim.x + threadIdx.x;
  long long tot = (long long)Np * Kp;
  if (idx >= tot) return;
  int k = (int)(idx % Kp);
  int n = (int)(idx / Kp);
  int K = cin * ks * ks;
  int N = Hout * Wout;
  float v = 0.f;
  if (n < N && k < K) {
    int c = k / (ks * ks);
    int r = k % (ks * ks);
    int kh = r / ks, kw = r % ks;
    int oy = n / Wout, ox = n % Wout;
    int y = oy - pad + kh * dil;
    int x = ox - pad + kw * dil;
    if (y >= 0 && y < Hin && x >= 0 && x < Win)
      v = act[((long long)c * Hin + y) * Win + x];
  }
  out[idx] = (_Float16)v;
}

// ---------------- transpose+convert: f32 [D][N] -> f16 [Np][Kp] ------------
__global__ void k_transpose_f16(const float* __restrict__ in, _Float16* __restrict__ out,
                                int D, int N, int Kp, int Np) {
  long long idx = (long long)blockIdx.x * blockDim.x + threadIdx.x;
  long long tot = (long long)Np * Kp;
  if (idx >= tot) return;
  int d = (int)(idx % Kp);
  int n = (int)(idx / Kp);
  float v = (d < D && n < N) ? in[(long long)d * N + n] : 0.f;
  out[idx] = (_Float16)v;
}

// ---------------- WMMA GEMM: C[M][N] = A[Mp][Kp] * Bt[Np][Kp]^T (+bias) ----
// 128 threads = 4 waves; each wave owns a 16x64 strip (4 N-tiles, NT=4).
// Np must be a multiple of 64 so all fragment loads are in-bounds.
#define GEMM_NT 4
__global__ void k_gemm_wmma(const _Float16* __restrict__ A,
                            const _Float16* __restrict__ Bt,
                            const float* __restrict__ bias,
                            float* __restrict__ C,
                            int M, int N, int Kp, int Np) {
  const int lane  = threadIdx.x & 31;
  const int wave  = threadIdx.x >> 5;
  const int group = blockIdx.x * (blockDim.x >> 5) + wave;  // 64-wide N group
  const int mtile = blockIdx.y;
  if (group * 64 >= Np) return;  // wave-uniform; EXEC all-ones at WMMA
  const int half = lane >> 4;
  const int l16  = lane & 15;
  const _Float16* Arow = A + (long long)(mtile * 16 + l16) * Kp;
  const _Float16* Bbase = Bt + (long long)(group * 64 + l16) * Kp;
  const long long Bstep = (long long)16 * Kp;  // one N-tile of rows
  v8f acc[GEMM_NT];
#pragma unroll
  for (int j = 0; j < GEMM_NT; ++j) acc[j] = (v8f){};
  for (int k0 = 0; k0 < Kp; k0 += 32) {
    union { v16h h; unsigned int u[8]; } a, b[GEMM_NT];
    const unsigned int* ap = (const unsigned int*)(Arow + k0);
#pragma unroll
    for (int v = 0; v < 8; ++v) {
      // A 16x32: lane m=l16; VGPR v holds K pair at (v/4)*16 + half*8 + (v%4)*2
      a.u[v] = ap[(v >> 2) * 8 + half * 4 + (v & 3)];
    }
#pragma unroll
    for (int j = 0; j < GEMM_NT; ++j) {
      const unsigned int* bp = (const unsigned int*)(Bbase + j * Bstep + k0);
#pragma unroll
      for (int v = 0; v < 8; ++v) {
        // B 32x16 (via Bt[n][k]): lane n=l16; VGPR v holds K pair at (v/4)*16 + half*8 + (v%4)*2
        b[j].u[v] = bp[(v >> 2) * 8 + half * 4 + (v & 3)];
      }
    }
    if (k0 + 32 < Kp) {
      __builtin_prefetch(Arow + k0 + 64, 0, 0);          // global_prefetch_b8
      __builtin_prefetch(Bbase + k0 + 64, 0, 0);
      __builtin_prefetch(Bbase + 2 * Bstep + k0 + 64, 0, 0);
    }
#pragma unroll
    for (int j = 0; j < GEMM_NT; ++j) {
      acc[j] = __builtin_amdgcn_wmma_f32_16x16x32_f16(false, a.h, false, b[j].h,
                                                      (short)0, acc[j], false, false);
    }
  }
#pragma unroll
  for (int j = 0; j < GEMM_NT; ++j) {
#pragma unroll
    for (int r = 0; r < 8; ++r) {
      int m = mtile * 16 + r + 8 * half;  // C/D layout: VGPR r -> row r (+8 hi half)
      int n = group * 64 + j * 16 + l16;
      if (m < M && n < N) {
        float v = acc[j][r];
        if (bias) v += bias[m];
        C[(long long)m * N + n] = v;
      }
    }
  }
}

// ---------------- BatchNorm(affine=False, batch stats) + ReLU, in place ----
__global__ void k_bn_relu(float* __restrict__ x, int N) {
  const int c = blockIdx.x;
  float* p = x + (long long)c * N;
  float s = 0.f, ss = 0.f;
  for (int i = threadIdx.x; i < N; i += blockDim.x) {
    float v = p[i];
    s += v; ss += v * v;
  }
  for (int off = 16; off > 0; off >>= 1) {
    s  += __shfl_down(s, off);
    ss += __shfl_down(ss, off);
  }
  __shared__ float sm0[16], sm1[16];
  int lane = threadIdx.x & 31, wv = threadIdx.x >> 5;
  if (lane == 0) { sm0[wv] = s; sm1[wv] = ss; }
  __syncthreads();
  if (threadIdx.x == 0) {
    int nw = blockDim.x >> 5;
    float ts = 0.f, tss = 0.f;
    for (int i = 0; i < nw; ++i) { ts += sm0[i]; tss += sm1[i]; }
    float mean = ts / (float)N;
    float var  = tss / (float)N - mean * mean;
    sm0[0] = mean;
    sm1[0] = rsqrtf(var + 1e-5f);
  }
  __syncthreads();
  float mean = sm0[0], inv = sm1[0];
  for (int i = threadIdx.x; i < N; i += blockDim.x) {
    float v = (p[i] - mean) * inv;
    p[i] = v > 0.f ? v : 0.f;
  }
}

// ---------------- 2x2 / s2 max-pool ----------------------------------------
__global__ void k_maxpool2(const float* __restrict__ in, float* __restrict__ out,
                           int C, int Hin, int Win) {
  int Ho = Hin >> 1, Wo = Win >> 1;
  long long idx = (long long)blockIdx.x * blockDim.x + threadIdx.x;
  long long tot = (long long)C * Ho * Wo;
  if (idx >= tot) return;
  int x = (int)(idx % Wo);
  long long t = idx / Wo;
  int y = (int)(t % Ho);
  int c = (int)(t / Ho);
  const float* b = in + ((long long)c * Hin + 2 * y) * Win + 2 * x;
  out[idx] = fmaxf(fmaxf(b[0], b[1]), fmaxf(b[Win], b[Win + 1]));
}

// ---------------- bilinear upsample (align_corners) + sigmoid --------------
__global__ void k_upsample_sigmoid(const float* __restrict__ in, float* __restrict__ unary,
                                   int C, int hin, int win, int oh, int ow) {
  long long idx = (long long)blockIdx.x * blockDim.x + threadIdx.x;
  long long tot = (long long)C * oh * ow;
  if (idx >= tot) return;
  int x = (int)(idx % ow);
  long long t = idx / ow;
  int y = (int)(t % oh);
  int c = (int)(t / oh);
  float fy = (float)y * (float)(hin - 1) / (float)(oh - 1);
  float fx = (float)x * (float)(win - 1) / (float)(ow - 1);
  int y0 = (int)floorf(fy); int y1 = min(y0 + 1, hin - 1);
  int x0 = (int)floorf(fx); int x1 = min(x0 + 1, win - 1);
  float wy = fy - (float)y0, wx = fx - (float)x0;
  const float* b = in + (long long)c * hin * win;
  float v00 = b[y0 * win + x0], v01 = b[y0 * win + x1];
  float v10 = b[y1 * win + x0], v11 = b[y1 * win + x1];
  float top = v00 * (1.f - wx) + v01 * wx;
  float bot = v10 * (1.f - wx) + v11 * wx;
  float v = top * (1.f - wy) + bot * wy;
  unary[idx] = 1.f / (1.f + __expf(-v));
}

// ---------------- color-affinity locally-connected layer -------------------
// One workgroup (8 waves) per output pixel; shuffle + LDS reductions.
__global__ void k_local(const float* __restrict__ input, const float* __restrict__ unary,
                        const float* __restrict__ loc_w, const float* __restrict__ loc_b,
                        float* __restrict__ out /* [21][4096] */) {
  const int p = blockIdx.x;           // 0..4095
  const int h = p >> 6, w = p & 63;
  const float i0 = input[p], i1 = input[4096 + p], i2 = input[2 * 4096 + p];
  float acc[NLAB];
#pragma unroll
  for (int c = 0; c < NLAB; ++c) acc[c] = 0.f;
  const float* lw = loc_w + (long long)p * KLOC2;
  for (int k = threadIdx.x; k < KLOC2; k += blockDim.x) {
    int dy = k / KLOC, dx = k - dy * KLOC;
    int y = h - 25 + dy, x = w - 25 + dx;
    if ((unsigned)y < 64u && (unsigned)x < 64u) {  // zero patches outside -> skip
      int q = (y << 6) | x;
      float d0 = i0 - input[q];
      float d1 = i1 - input[4096 + q];
      float d2 = i2 - input[2 * 4096 + q];
      float sw = lw[k] * __expf(-(d0 * d0 + d1 * d1 + d2 * d2));
      const float* U = unary + q;
#pragma unroll
      for (int c = 0; c < NLAB; ++c) acc[c] += sw * U[c * 4096];
    }
  }
  __shared__ float red[16];
  const int lane = threadIdx.x & 31, wv = threadIdx.x >> 5;
  const int nw = blockDim.x >> 5;
  for (int c = 0; c < NLAB; ++c) {
    float v = acc[c];
    for (int off = 16; off > 0; off >>= 1) v += __shfl_down(v, off);
    if (lane == 0) red[wv] = v;
    __syncthreads();
    if (threadIdx.x == 0) {
      float t = 0.f;
      for (int i = 0; i < nw; ++i) t += red[i];
      out[(long long)c * 4096 + p] = t + loc_b[c];
    }
    __syncthreads();
  }
}

// ---------------- small pixel-wise linear: out[d][p] = W[d][:] . in[:][p] ---
__global__ void k_lin(const float* __restrict__ Wm, const float* __restrict__ b,
                      const float* __restrict__ in, float* __restrict__ out,
                      int D, int Kc, int N) {
  long long idx = (long long)blockIdx.x * blockDim.x + threadIdx.x;
  long long tot = (long long)D * N;
  if (idx >= tot) return;
  int p = (int)(idx % N);
  int d = (int)(idx / N);
  float s = b[d];
  for (int c = 0; c < Kc; ++c) s += Wm[(long long)d * Kc + c] * in[(long long)c * N + p];
  out[idx] = s;
}

// ---------------- block-min over mixtures + log(unary)-smooth + log_softmax -
__global__ void k_final(const float* __restrict__ unary, const float* __restrict__ y4,
                        float* __restrict__ out) {
  int p = blockIdx.x * blockDim.x + threadIdx.x;
  if (p >= 4096) return;
  float logit[NLAB];
  float mx = -1e30f;
#pragma unroll
  for (int c = 0; c < NLAB; ++c) {
    float s = 1e30f;
#pragma unroll
    for (int m = 0; m < NMIX; ++m)
      s = fminf(s, y4[(long long)(m * NLAB + c) * 4096 + p]);
    float l = __logf(unary[(long long)c * 4096 + p]) - s;
    logit[c] = l;
    mx = fmaxf(mx, l);
  }
  float se = 0.f;
#pragma unroll
  for (int c = 0; c < NLAB; ++c) se += __expf(logit[c] - mx);
  float lse = mx + __logf(se);
#pragma unroll
  for (int c = 0; c < NLAB; ++c) out[(long long)c * 4096 + p] = logit[c] - lse;
}

// ======================= host orchestration ================================
static void launch_gemm(const _Float16* packA, const _Float16* packB,
                        const float* bias, float* C,
                        int M, int N, int Kp, int Np, hipStream_t stream) {
  int groups = Np / 64;                       // 64-wide N groups
  dim3 g((unsigned)((groups + 3) / 4), (unsigned)(ceil_to(M, 16) / 16));
  k_gemm_wmma<<<g, 128, 0, stream>>>(packA, packB, bias, C, M, N, Kp, Np);
}

static void run_conv(const float* actIn, float* actOut,
                     const float* Wf, const float* Bf,
                     int cin, int cout, int ks, int dil, int pad,
                     int Hin, int Win, int* Hout, int* Wout,
                     _Float16* packA, _Float16* packB, hipStream_t stream) {
  int Ho = Hin + 2 * pad - dil * (ks - 1);
  int Wo = Win + 2 * pad - dil * (ks - 1);
  *Hout = Ho; *Wout = Wo;
  int K  = cin * ks * ks;
  int Kp = ceil_to(K, 32);
  int M  = cout, Mp = ceil_to(M, 16);
  int N  = Ho * Wo, Np = ceil_to(N, 64);
  long long tA = (long long)Mp * Kp;
  k_pack_w<<<(unsigned)((tA + 255) / 256), 256, 0, stream>>>(Wf, packA, M, K, Mp, Kp);
  long long tB = (long long)Np * Kp;
  k_im2col<<<(unsigned)((tB + 255) / 256), 256, 0, stream>>>(
      actIn, packB, cin, Hin, Win, Ho, Wo, ks, dil, pad, Kp, Np);
  launch_gemm(packA, packB, Bf, actOut, M, N, Kp, Np, stream);
  k_bn_relu<<<cout, 256, 0, stream>>>(actOut, N);
}

extern "C" void kernel_launch(void* const* d_in, const int* in_sizes, int n_in,
                              void* d_out, int out_size, void* d_ws, size_t ws_size,
                              hipStream_t stream) {
  (void)in_sizes; (void)n_in; (void)out_size; (void)ws_size;
  const float* input = (const float*)d_in[0];          // [3][64][64]
  const float* vgg_w[16];
  const float* vgg_b[16];
  for (int i = 0; i < 16; ++i) {
    vgg_w[i] = (const float*)d_in[1 + i];
    vgg_b[i] = (const float*)d_in[17 + i];
  }
  const float* loc_w  = (const float*)d_in[33];
  const float* loc_b  = (const float*)d_in[34];
  const float* lin1_w = (const float*)d_in[35];
  const float* lin1_b = (const float*)d_in[36];
  const float* glob_w = (const float*)d_in[37];
  const float* glob_b = (const float*)d_in[38];
  const float* lin2_w = (const float*)d_in[39];
  const float* lin2_b = (const float*)d_in[40];
  float* out = (float*)d_out;

  char* ws = (char*)d_ws;
  size_t off = 0;
  auto alloc = [&](size_t bytes) -> void* {
    void* p = (void*)(ws + off);
    off += (bytes + 255) & ~(size_t)255;
    return p;
  };
  _Float16* packA = (_Float16*)alloc((size_t)4096 * 25088 * sizeof(_Float16)); // 205.5 MB
  _Float16* packB = (_Float16*)alloc((size_t)4096 * 1984 * sizeof(_Float16));  // 16.3 MB
  float*    actA  = (float*)alloc((size_t)1 << 22);                            // 4 MB
  float*    actB  = (float*)alloc((size_t)1 << 22);                            // 4 MB
  float*    unary = (float*)alloc((size_t)NLAB * 4096 * sizeof(float));

  struct LC { int cin, cout, ks, dil, pad; };
  static const LC conv[16] = {
    {3, 64, 3, 1, 1},   {64, 64, 3, 1, 1},
    {64, 128, 3, 1, 1}, {128, 128, 3, 1, 1},
    {128, 256, 3, 1, 1},{256, 256, 3, 1, 1}, {256, 256, 3, 1, 1},
    {256, 512, 3, 1, 1},{512, 512, 3, 1, 1}, {512, 512, 3, 1, 1},
    {512, 512, 3, 2, 2},{512, 512, 3, 2, 2}, {512, 512, 3, 2, 2},
    {512, 4096, 7, 4, 12},
    {4096, 4096, 1, 1, 1},
    {4096, NLAB, 1, 1, 0},
  };

  // -------- VGG backbone --------
  int H = 64, W = 64;
  const float* cur = input;
  float* bufs[2] = {actA, actB};
  int pi = 0;
  for (int ci = 0; ci < 16; ++ci) {
    float* outBuf = bufs[pi];
    int Ho, Wo;
    run_conv(cur, outBuf, vgg_w[ci], vgg_b[ci],
             conv[ci].cin, conv[ci].cout, conv[ci].ks, conv[ci].dil, conv[ci].pad,
             H, W, &Ho, &Wo, packA, packB, stream);
    H = Ho; W = Wo;
    cur = outBuf; pi ^= 1;
    if (ci == 1 || ci == 3 || ci == 6) {
      float* poolOut = bufs[pi];
      long long tot = (long long)conv[ci].cout * (H / 2) * (W / 2);
      k_maxpool2<<<(unsigned)((tot + 255) / 256), 256, 0, stream>>>(
          cur, poolOut, conv[ci].cout, H, W);
      H /= 2; W /= 2;
      cur = poolOut; pi ^= 1;
    }
  }
  // cur = [21][10][10] -> unary [21][64][64]
  {
    long long tot = (long long)NLAB * 64 * 64;
    k_upsample_sigmoid<<<(unsigned)((tot + 255) / 256), 256, 0, stream>>>(
        cur, unary, NLAB, H, W, 64, 64);
  }

  // -------- color-affinity locally-connected layer -> actA [21][4096] -----
  k_local<<<4096, 256, 0, stream>>>(input, unary, loc_w, loc_b, actA);

  // -------- lin1 (21x21 per-pixel) -> actB [21][4096] ----------------------
  {
    long long tot = (long long)NLAB * 4096;
    k_lin<<<(unsigned)((tot + 255) / 256), 256, 0, stream>>>(
        lin1_w, lin1_b, actA, actB, NLAB, NLAB, 4096);
  }

  // -------- glob 9x9 conv (21 -> 210), pad 4, via WMMA GEMM -> actA --------
  {
    int K = NLAB * 81, Kp = ceil_to(K, 32);          // 1701 -> 1728
    int M = NLAB * NMIX;                             // 210
    int Mp = ceil_to(M, 16);                         // 224
    int N = 4096, Np = 4096;
    long long tA = (long long)Mp * Kp;
    k_pack_w<<<(unsigned)((tA + 255) / 256), 256, 0, stream>>>(glob_w, packA, M, K, Mp, Kp);
    long long tB = (long long)Np * Kp;
    k_im2col<<<(unsigned)((tB + 255) / 256), 256, 0, stream>>>(
        actB, packB, NLAB, 64, 64, 64, 64, 9, 1, 4, Kp, Np);
    launch_gemm(packA, packB, glob_b, actA, M, N, Kp, Np, stream);
  }

  // -------- lin2 (210x210 per-pixel) via WMMA GEMM -> actB -----------------
  {
    int D = NLAB * NMIX;                              // 210
    int Kp = ceil_to(D, 32);                          // 224
    int Mp = ceil_to(D, 16);                          // 224
    int N = 4096, Np = 4096;
    long long tT = (long long)Np * Kp;
    k_transpose_f16<<<(unsigned)((tT + 255) / 256), 256, 0, stream>>>(
        actA, packB, D, N, Kp, Np);
    long long tA = (long long)Mp * Kp;
    k_pack_w<<<(unsigned)((tA + 255) / 256), 256, 0, stream>>>(lin2_w, packA, D, D, Mp, Kp);
    launch_gemm(packA, packB, lin2_b, actB, D, N, Kp, Np, stream);
  }

  // -------- block-min + log(unary) - smooth + log_softmax -> out -----------
  k_final<<<16, 256, 0, stream>>>(unary, actB, out);
}
